// DPLSTMCell_37915971289787
// MI455X (gfx1250) — matrix operand order repaired and checked by reading, not compiled, not run on hardware
//
#include <hip/hip_runtime.h>
#include <stdint.h>

// ---------------------------------------------------------------------------
// LSTM cell for MI455X (gfx1250, wave32)
//   gates[B,4H] = X[B,2048] @ Wc[4H,2048]^T + b   via v_wmma_f32_16x16x32_f16
//   global->LDS movement via GLOBAL_LOAD_ASYNC_TO_LDS_B128 (ASYNCcnt path)
//   B fragments software-pipelined in registers to hide LDS latency
//   fused epilogue: c1 = sig(f)*c0 + sig(i)*tanh(g); h1 = sig(o)*tanh(c1)
// ---------------------------------------------------------------------------

typedef __attribute__((ext_vector_type(16))) _Float16 v16h;
typedef __attribute__((ext_vector_type(8)))  _Float16 v8h;
typedef __attribute__((ext_vector_type(4)))  _Float16 v4h;
typedef __attribute__((ext_vector_type(8)))  float    v8f;

#define B_SZ   16384
#define H_SZ   1024
#define IN_SZ  1024
#define K_TOT  2048      // IN + H concatenated along K
#define M_BLK  256       // rows per block (8 waves x 32 rows)
#define N_BLK  32        // h-columns per block (x4 gates computed together)
#define K_STEP 32
#define LDS_LD 40        // f16 per LDS row: 32 + 8 pad (80B: bank-spread, 16B-aligned chunks)
#define A_ROWS M_BLK
#define B_ROWS 128       // 4 gates * 32 n
#define NSTEP  (K_TOT / K_STEP)

__device__ __forceinline__ float sigmoid_f(float x) {
    return 1.0f / (1.0f + __expf(-x));
}

// CDNA5 async global->LDS copy, 16B per lane, tracked by ASYNCcnt.
__device__ __forceinline__ void async_load_b128(uint32_t lds_addr, const void* gptr) {
    asm volatile("global_load_async_to_lds_b128 %0, %1, off"
                 :: "v"(lds_addr), "v"(gptr)
                 : "memory");
}
__device__ __forceinline__ void wait_async0() {
    asm volatile("s_wait_asynccnt 0x0" ::: "memory");
}

// ------------------- f32 -> f16 pack into strided layout --------------------
__global__ __launch_bounds__(256) void cvt_pack_f16(
    const float* __restrict__ src, _Float16* __restrict__ dst,
    int src_cols, int dst_stride, int dst_off, long total_vec4)
{
    long i = (long)blockIdx.x * blockDim.x + threadIdx.x;
    if (i >= total_vec4) return;
    long e = i * 4;
    long r = e / src_cols;
    int  c = (int)(e - r * (long)src_cols);
    const float4 v = *(const float4*)(src + e);
    v4h o;
    o[0] = (_Float16)v.x; o[1] = (_Float16)v.y;
    o[2] = (_Float16)v.z; o[3] = (_Float16)v.w;
    *(v4h*)(dst + r * (long)dst_stride + dst_off + c) = o;
}

// ----------------------------- fused LSTM GEMM ------------------------------
__global__ __launch_bounds__(256) void lstm_wmma_kernel(
    const _Float16* __restrict__ Xh,   // [16384, 2048] f16 (input | h_0)
    const _Float16* __restrict__ Wc,   // [4096, 2048]  f16 (W_ih | W_hh), row = g*1024 + h
    const float* __restrict__ b_ih,    // [4096]
    const float* __restrict__ b_hh,    // [4096]
    const float* __restrict__ c0,      // [16384, 1024]
    float* __restrict__ h1,            // [16384, 1024]
    float* __restrict__ c1)            // [16384, 1024]
{
    __shared__ __attribute__((aligned(16))) _Float16 As[2][A_ROWS][LDS_LD]; // 40 KB
    __shared__ __attribute__((aligned(16))) _Float16 Bs[2][B_ROWS][LDS_LD]; // 20 KB

    const int tid   = threadIdx.x;
    const int wave  = tid >> 5;
    const int lane  = tid & 31;
    const int lhalf = lane >> 4;     // 0: lanes 0-15, 1: lanes 16-31
    const int l16   = lane & 15;
    const int mBase = blockIdx.y * M_BLK;
    const int nBase = blockIdx.x * N_BLK;

    // acc[mt][gate][nt] : each wave owns 2 m-tiles -> B fragments reused 2x
    v8f acc[2][4][2];
#pragma unroll
    for (int mt = 0; mt < 2; ++mt)
#pragma unroll
        for (int g = 0; g < 4; ++g)
#pragma unroll
            for (int nt = 0; nt < 2; ++nt)
#pragma unroll
                for (int e = 0; e < 8; ++e) acc[mt][g][nt][e] = 0.0f;

    // ---- per-thread global->LDS chunk map (16B chunks) --------------------
    // A tile: 256 rows x 4 chunks = 1024 chunks -> 4 per thread
    // B tile: 128 rows x 4 chunks =  512 chunks -> 2 per thread
    const _Float16* agp[4];
    uint32_t aoff[4];
#pragma unroll
    for (int i = 0; i < 4; ++i) {
        const int c  = tid + i * 256;
        const int rw = c >> 2, cc = (c & 3) * 8;
        agp[i]  = Xh + (size_t)(mBase + rw) * K_TOT + cc;
        aoff[i] = (uint32_t)(rw * LDS_LD + cc) * 2;
    }
    const _Float16* bgp[2];
    uint32_t boff[2];
#pragma unroll
    for (int i = 0; i < 2; ++i) {
        const int c  = tid + i * 256;
        const int j  = c >> 2, cc = (c & 3) * 8;
        const int wr = (j >> 5) * H_SZ + nBase + (j & 31);
        bgp[i]  = Wc + (size_t)wr * K_TOT + cc;
        boff[i] = (uint32_t)(j * LDS_LD + cc) * 2;
    }

    // LDS byte addresses: low 32 bits of the generic LDS pointer
    const uint32_t aBase = (uint32_t)(uintptr_t)(&As[0][0][0]);
    const uint32_t bBase = (uint32_t)(uintptr_t)(&Bs[0][0][0]);
    const uint32_t aBufB = A_ROWS * LDS_LD * 2;  // 20480
    const uint32_t bBufB = B_ROWS * LDS_LD * 2;  // 10240

    // ---- prologue: fill buffer 0 asynchronously ---------------------------
#pragma unroll
    for (int i = 0; i < 4; ++i) async_load_b128(aBase + aoff[i], agp[i]);
#pragma unroll
    for (int i = 0; i < 2; ++i) async_load_b128(bBase + boff[i], bgp[i]);
    wait_async0();
    __syncthreads();

    for (int ks = 0; ks < NSTEP; ++ks) {
        const int buf = ks & 1;

        // issue async loads for the next k-slice into the other buffer;
        // they overlap the 16 WMMAs below and are waited on before the barrier
        if (ks + 1 < NSTEP) {
            const int k0 = (ks + 1) * K_STEP;
            const uint32_t nb = (uint32_t)(buf ^ 1);
#pragma unroll
            for (int i = 0; i < 4; ++i)
                async_load_b128(aBase + nb * aBufB + aoff[i], agp[i] + k0);
#pragma unroll
            for (int i = 0; i < 2; ++i)
                async_load_b128(bBase + nb * bBufB + boff[i], bgp[i] + k0);
        }

        // A fragments (16x32 f16): lane<16 holds row m=l16, K {0..7,16..23};
        // lane>=16 same row, K {8..15,24..31}.
        v16h afrag[2];
#pragma unroll
        for (int mt = 0; mt < 2; ++mt) {
            const _Float16* Ar = &As[buf][wave * 32 + mt * 16 + l16][lhalf * 8];
            v8h alo = *(const v8h*)(Ar);
            v8h ahi = *(const v8h*)(Ar + 16);
#pragma unroll
            for (int e = 0; e < 8; ++e) { afrag[mt][e] = alo[e]; afrag[mt][e + 8] = ahi[e]; }
        }

        // B fragment loader: (32x16 f16) lane holds column n=l16, K {lhalf*8 + 0..7, 16 + lhalf*8 + 0..7}
        auto loadB = [&](int f) -> v16h {
            const int g = f >> 1, nt = f & 1;
            const _Float16* Br = &Bs[buf][g * 32 + nt * 16 + l16][lhalf * 8];
            v8h blo = *(const v8h*)(Br);
            v8h bhi = *(const v8h*)(Br + 16);
            v16h b;
#pragma unroll
            for (int e = 0; e < 8; ++e) { b[e] = blo[e]; b[e + 8] = bhi[e]; }
            return b;
        };

        // 1-deep register pipeline over the 8 B fragments: loads for f+1 are
        // in flight while the two WMMAs of f execute (relaxes s_wait_dscnt)
        v16h bcur = loadB(0);
#pragma unroll
        for (int f = 0; f < 8; ++f) {
            v16h bnext;
            if (f < 7) bnext = loadB(f + 1);
            const int g = f >> 1, nt = f & 1;
            acc[0][g][nt] = __builtin_amdgcn_wmma_f32_16x16x32_f16(
                false, afrag[0], false, bcur, (short)0, acc[0][g][nt], false, false);
            acc[1][g][nt] = __builtin_amdgcn_wmma_f32_16x16x32_f16(
                false, afrag[1], false, bcur, (short)0, acc[1][g][nt], false, false);
            if (f < 7) bcur = bnext;
        }

        wait_async0();      // our async writes to LDS have landed
        __syncthreads();    // publish to all waves; also fences reads of 'buf'
    }

    // ---- fused LSTM epilogue --------------------------------------------
    // D layout (16x16 f32): VGPR r -> lanes 0-15: (M=r, N=l16); lanes 16-31: (M=r+8, N=l16)
#pragma unroll
    for (int nt = 0; nt < 2; ++nt) {
        const int nh = nBase + nt * 16 + l16;
        const float biasI = b_ih[0 * H_SZ + nh] + b_hh[0 * H_SZ + nh];
        const float biasF = b_ih[1 * H_SZ + nh] + b_hh[1 * H_SZ + nh];
        const float biasG = b_ih[2 * H_SZ + nh] + b_hh[2 * H_SZ + nh];
        const float biasO = b_ih[3 * H_SZ + nh] + b_hh[3 * H_SZ + nh];
#pragma unroll
        for (int mt = 0; mt < 2; ++mt) {
            const int mW = mBase + wave * 32 + mt * 16;
#pragma unroll
            for (int r = 0; r < 8; ++r) {
                const int m = mW + r + lhalf * 8;
                const float gi = sigmoid_f(acc[mt][0][nt][r] + biasI);
                const float gf = sigmoid_f(acc[mt][1][nt][r] + biasF);
                const float gg = tanhf   (acc[mt][2][nt][r] + biasG);
                const float go = sigmoid_f(acc[mt][3][nt][r] + biasO);
                const size_t idx = (size_t)m * H_SZ + nh;
                const float cv = gf * c0[idx] + gi * gg;
                c1[idx] = cv;
                h1[idx] = go * tanhf(cv);
            }
        }
    }
}

// ---------------------------------------------------------------------------
extern "C" void kernel_launch(void* const* d_in, const int* in_sizes, int n_in,
                              void* d_out, int out_size, void* d_ws, size_t ws_size,
                              hipStream_t stream) {
    const float* input = (const float*)d_in[0];   // [16384,1024]
    const float* h_0   = (const float*)d_in[1];   // [16384,1024]
    const float* c_0   = (const float*)d_in[2];   // [16384,1024]
    const float* W_ih  = (const float*)d_in[3];   // [4096,1024]
    const float* b_ih  = (const float*)d_in[4];   // [4096]
    const float* W_hh  = (const float*)d_in[5];   // [4096,1024]
    const float* b_hh  = (const float*)d_in[6];   // [4096]

    _Float16* Xh = (_Float16*)d_ws;                         // [16384,2048] = 64 MB
    _Float16* Wc = Xh + (size_t)B_SZ * K_TOT;               // [4096,2048]  = 16 MB
    float* h1 = (float*)d_out;
    float* c1 = h1 + (size_t)B_SZ * H_SZ;

    // f32 -> f16 packing (K-contiguous concat layout; X+W fit in 192MB L2)
    {
        long t;
        t = (long)B_SZ * IN_SZ / 4;
        cvt_pack_f16<<<(t + 255) / 256, 256, 0, stream>>>(input, Xh, IN_SZ, K_TOT, 0, t);
        t = (long)B_SZ * H_SZ / 4;
        cvt_pack_f16<<<(t + 255) / 256, 256, 0, stream>>>(h_0, Xh, H_SZ, K_TOT, IN_SZ, t);
        t = (long)4 * H_SZ * IN_SZ / 4;
        cvt_pack_f16<<<(t + 255) / 256, 256, 0, stream>>>(W_ih, Wc, IN_SZ, K_TOT, 0, t);
        t = (long)4 * H_SZ * H_SZ / 4;
        cvt_pack_f16<<<(t + 255) / 256, 256, 0, stream>>>(W_hh, Wc, H_SZ, K_TOT, IN_SZ, t);
    }

    dim3 grid(H_SZ / N_BLK, B_SZ / M_BLK);   // 32 x 64
    lstm_wmma_kernel<<<grid, 256, 0, stream>>>(Xh, Wc, b_ih, b_hh, c_0, h1, c1);
}
